// MultiHeadAttention_37933151158277
// MI455X (gfx1250) — matrix operand order repaired and checked by reading, not compiled, not run on hardware
//
#include <hip/hip_runtime.h>
#include <hip/hip_bf16.h>

// ---------------- CDNA5 WMMA types ----------------
typedef __attribute__((ext_vector_type(16))) _Float16 v16h;
typedef __attribute__((ext_vector_type(8)))  _Float16 v8h;
typedef __attribute__((ext_vector_type(8)))  float    v8f;

union Frag16 { v16h v; v8h h[2]; };

// A-matrix (16x32 f16) per-lane layout (ISA 7.12.2):
//   lanes 0-15 : row M=lane,     halves K = [0..7] , [16..23]
//   lanes16-31 : row M=lane-16,  halves K = [8..15], [24..31]
__device__ inline v16h load_a_frag(const _Float16* base, int ld, int m0, int k0, int lane) {
    const _Float16* row = base + (size_t)(m0 + (lane & 15)) * ld;
    int kh = k0 + ((lane >> 4) << 3);
    Frag16 fr;
    fr.h[0] = *(const v8h*)(row + kh);
    fr.h[1] = *(const v8h*)(row + kh + 16);
    return fr.v;
}
// B-matrix (32x16 f16): lanes 0-15 col N=lane, K=[0..7],[16..23]; lanes 16-31 K=[8..15],[24..31].
// Stored as [N][K] (i.e. B^T row-major), so a column of B is a contiguous row here.
__device__ inline v16h load_b_frag(const _Float16* base, int ld, int n0, int k0, int lane) {
    const _Float16* row = base + (size_t)(n0 + (lane & 15)) * ld;
    int kh = k0 + ((lane >> 4) << 3);
    Frag16 fr;
    fr.h[0] = *(const v8h*)(row + kh);
    fr.h[1] = *(const v8h*)(row + kh + 16);
    return fr.v;
}

// -------- CDNA5 async global->LDS copy (ASYNCcnt-tracked, bypasses VGPRs) --------
// VGLOBAL encoding: VDST = LDS byte address (addr[31:0] of shared aperture),
// VADDR = 64-bit global address, GV mode (saddr=off).
__device__ inline void async_copy_b128(void* lds_dst, const void* gsrc) {
    unsigned       lds_off = (unsigned)(unsigned long long)lds_dst;
    unsigned long long ga  = (unsigned long long)gsrc;
    asm volatile("global_load_async_to_lds_b128 %0, %1, off"
                 :: "v"(lds_off), "v"(ga) : "memory");
}
__device__ inline void async_wait0() {
    asm volatile("s_wait_asynccnt 0x0" ::: "memory");
}

// ---------------- fp32 -> f16 convert ----------------
__global__ void cvt_f32_f16(const float* __restrict__ in, _Float16* __restrict__ out, int n) {
    for (int i = blockIdx.x * blockDim.x + threadIdx.x; i < n; i += gridDim.x * blockDim.x)
        out[i] = (_Float16)in[i];
}

// ---------------- NT GEMM: C[M,N] = A[M,K] * B[N,K]^T ----------------
// Block tile 128x128x32, 256 threads = 8 waves, each wave 32(m) x 64(n).
#define GBM 128
#define GBN 128
#define GBK 32
#define GLD 40   // padded LDS row stride (halves); 20 dwords -> conflict-free

template <bool OUTF32>
__global__ __launch_bounds__(256) void wmma_gemm_nt(
    const _Float16* __restrict__ A, const _Float16* __restrict__ B,
    void* __restrict__ C, int M, int N, int K)
{
    __shared__ _Float16 As[GBM * GLD];
    __shared__ _Float16 Bs[GBN * GLD];

    const int tid  = threadIdx.x;
    const int lane = tid & 31;
    const int wave = tid >> 5;
    const int wm   = wave & 3;   // 4 m-blocks of 32 rows
    const int wn   = wave >> 2;  // 2 n-blocks of 64 cols

    const int mBlock = blockIdx.y * GBM;
    const int nBlock = blockIdx.x * GBN;

    v8f acc[2][4] = {};

    const int loadRow = tid >> 1;          // 0..127
    const int loadCol = (tid & 1) << 4;    // 0 or 16 halves
    const _Float16* aptr = A + (size_t)(mBlock + loadRow) * K + loadCol;
    const _Float16* bptr = B + (size_t)(nBlock + loadRow) * K + loadCol;
    _Float16* asr = &As[loadRow * GLD + loadCol];
    _Float16* bsr = &Bs[loadRow * GLD + loadCol];

    for (int kt = 0; kt < K; kt += GBK) {
        // async tile staging: global -> LDS directly, no VGPR round-trip
        async_copy_b128(asr,     aptr + kt);
        async_copy_b128(asr + 8, aptr + kt + 8);
        async_copy_b128(bsr,     bptr + kt);
        async_copy_b128(bsr + 8, bptr + kt + 8);
        async_wait0();
        __syncthreads();

        v16h af[2], bf[4];
        #pragma unroll
        for (int i = 0; i < 2; ++i) af[i] = load_a_frag(As, GLD, wm * 32 + i * 16, 0, lane);
        #pragma unroll
        for (int j = 0; j < 4; ++j) bf[j] = load_b_frag(Bs, GLD, wn * 64 + j * 16, 0, lane);
        #pragma unroll
        for (int i = 0; i < 2; ++i)
            #pragma unroll
            for (int j = 0; j < 4; ++j)
                acc[i][j] = __builtin_amdgcn_wmma_f32_16x16x32_f16(
                    false, af[i], false, bf[j], (short)0, acc[i][j], false, false);
        __syncthreads();
    }

    // C/D layout: VGPR r -> M = r + 8*(lane>=16); N = lane&15
    #pragma unroll
    for (int i = 0; i < 2; ++i) {
        #pragma unroll
        for (int j = 0; j < 4; ++j) {
            const int rowBase = mBlock + wm * 32 + i * 16 + ((lane >> 4) << 3);
            const int col     = nBlock + wn * 64 + j * 16 + (lane & 15);
            #pragma unroll
            for (int r = 0; r < 8; ++r) {
                const size_t idx = (size_t)(rowBase + r) * N + col;
                if (OUTF32) ((float*)C)[idx] = acc[i][j][r];
                else        ((_Float16*)C)[idx] = (_Float16)acc[i][j][r];
            }
        }
    }
}

// ---------------- Flash attention (causal) ----------------
// One block handles (b, h, 32-query tile). Key tiles of 64.
// qkv rows: [B*S][6144], Q at h*128, K at 2048+h*128, V at 4096+h*128.
#define ATT_S   2048
#define ATT_LDQ 136   // 68 dwords/row -> stride 4 banks, conflict-free
#define ATT_LDV 72    // 36 dwords/row
#define ATT_LDS 68    // score rows, f32
#define ATT_LDP 72

__global__ __launch_bounds__(256) void flash_attn_kernel(
    const _Float16* __restrict__ qkv, _Float16* __restrict__ ctx)
{
    __shared__ _Float16 Qs[32 * ATT_LDQ];
    __shared__ _Float16 Ks[64 * ATT_LDQ];
    __shared__ _Float16 Vt[128 * ATT_LDV];   // transposed: [dk][key]
    __shared__ float    Sc[32 * ATT_LDS];
    __shared__ _Float16 Pp[32 * ATT_LDP];
    __shared__ float    scl[32];

    const int tid  = threadIdx.x;
    const int lane = tid & 31;
    const int wave = tid >> 5;
    const int q0 = blockIdx.x * 32;
    const int h  = blockIdx.y;
    const int b  = blockIdx.z;

    const size_t ldqkv = 6144;
    const _Float16* qbase = qkv + (size_t)(b * ATT_S) * ldqkv + h * 128;
    const _Float16* kbase = qbase + 2048;
    const _Float16* vbase = qbase + 4096;

    // Load Q tile once (async): thread t -> row t>>3 (0..31), seg (t&7)*16 halves
    {
        const int r = tid >> 3, seg = (tid & 7) << 4;
        const _Float16* g = qbase + (size_t)(q0 + r) * ldqkv + seg;
        _Float16* d = &Qs[r * ATT_LDQ + seg];
        async_copy_b128(d,     g);
        async_copy_b128(d + 8, g + 8);
    }

    float m_i = -1e30f, l_i = 0.f;     // live in threads 0..31 only
    v8f oacc[2] = {};
    const int smt = wave & 1, snt = wave >> 1;   // score tile (2m x 4n of 16x16)
    const int omt = wave & 1, onb = wave >> 1;   // O tile: rows omt*16, cols onb*32

    const int jend = (q0 + 31) >> 6;             // causal: last key tile touching q0..q0+31
    for (int j = 0; j <= jend; ++j) {
        // Load K tile [64][128] async: thread t -> row t>>2, seg (t&3)*32 halves
        {
            const int r = tid >> 2, seg = (tid & 3) << 5;
            const _Float16* g = kbase + (size_t)(j * 64 + r) * ldqkv + seg;
            _Float16* d = &Ks[r * ATT_LDQ + seg];
            async_copy_b128(d,      g);
            async_copy_b128(d + 8,  g + 8);
            async_copy_b128(d + 16, g + 16);
            async_copy_b128(d + 24, g + 24);
        }
        // Load V tile transposed (manual): thread t -> key t&63, dk seg (t>>6)*32
        {
            const int key = tid & 63, dk0 = (tid >> 6) << 5;
            const _Float16* g = vbase + (size_t)(j * 64 + key) * ldqkv + dk0;
            v8h tmp[4];
            tmp[0] = *(const v8h*)(g);      tmp[1] = *(const v8h*)(g + 8);
            tmp[2] = *(const v8h*)(g + 16); tmp[3] = *(const v8h*)(g + 24);
            #pragma unroll
            for (int c = 0; c < 4; ++c)
                #pragma unroll
                for (int e = 0; e < 8; ++e)
                    Vt[(dk0 + c * 8 + e) * ATT_LDV + key] = tmp[c][e];
        }
        async_wait0();   // covers K tile now and Q tile on first iteration
        __syncthreads();

        // scores: S = Q * K^T, tile 16x16 per wave, dk reduced in 4 WMMA steps
        v8f sacc = {};
        #pragma unroll
        for (int kk = 0; kk < 4; ++kk) {
            v16h aq = load_a_frag(Qs, ATT_LDQ, smt * 16, kk * 32, lane);
            v16h bk = load_b_frag(Ks, ATT_LDQ, snt * 16, kk * 32, lane);
            sacc = __builtin_amdgcn_wmma_f32_16x16x32_f16(
                false, aq, false, bk, (short)0, sacc, false, false);
        }
        // write scaled + causally-masked scores to LDS
        {
            const int colL = snt * 16 + (lane & 15);
            const int rowB = smt * 16 + ((lane >> 4) << 3);
            #pragma unroll
            for (int r = 0; r < 8; ++r) {
                const int row = rowB + r;
                const int kg = j * 64 + colL, qg = q0 + row;
                const float v = sacc[r] * 0.08838834764831845f;  // 1/sqrt(128)
                Sc[row * ATT_LDS + colL] = (kg <= qg) ? v : -1e30f;
            }
        }
        __syncthreads();

        // online softmax: one thread per query row
        if (tid < 32) {
            float tm = -1e30f;
            for (int c = 0; c < 64; ++c) tm = fmaxf(tm, Sc[tid * ATT_LDS + c]);
            const float nm = fmaxf(m_i, tm);
            const float corr = __expf(m_i - nm);
            float s = 0.f;
            for (int c = 0; c < 64; ++c) {
                const float p = __expf(Sc[tid * ATT_LDS + c] - nm);
                s += p;
                Pp[tid * ATT_LDP + c] = (_Float16)p;
            }
            l_i = l_i * corr + s;
            m_i = nm;
            scl[tid] = corr;
        }
        __syncthreads();

        // rescale running O, then O += P * V  (keys reduced in 2 WMMA steps)
        {
            const int rowB = omt * 16 + ((lane >> 4) << 3);
            float sc8[8];
            #pragma unroll
            for (int r = 0; r < 8; ++r) sc8[r] = scl[rowB + r];
            #pragma unroll
            for (int t = 0; t < 2; ++t)
                #pragma unroll
                for (int r = 0; r < 8; ++r) oacc[t][r] *= sc8[r];
            #pragma unroll
            for (int ks = 0; ks < 2; ++ks) {
                v16h ap = load_a_frag(Pp, ATT_LDP, omt * 16, ks * 32, lane);
                #pragma unroll
                for (int nt = 0; nt < 2; ++nt) {
                    v16h bv = load_b_frag(Vt, ATT_LDV, onb * 32 + nt * 16, ks * 32, lane);
                    oacc[nt] = __builtin_amdgcn_wmma_f32_16x16x32_f16(
                        false, ap, false, bv, (short)0, oacc[nt], false, false);
                }
            }
        }
        __syncthreads();
    }

    if (tid < 32) scl[tid] = 1.f / l_i;
    __syncthreads();

    // normalize and write context as f16: ctx[b*S+q][h*128 + dk]
    {
        const int rowB = omt * 16 + ((lane >> 4) << 3);
        float inv8[8];
        #pragma unroll
        for (int r = 0; r < 8; ++r) inv8[r] = scl[rowB + r];
        #pragma unroll
        for (int t = 0; t < 2; ++t) {
            const int col = h * 128 + onb * 32 + t * 16 + (lane & 15);
            #pragma unroll
            for (int r = 0; r < 8; ++r) {
                const int row = q0 + rowB + r;
                ctx[(size_t)(b * ATT_S + row) * 2048 + col] = (_Float16)(oacc[t][r] * inv8[r]);
            }
        }
    }
}

// ---------------- launcher ----------------
extern "C" void kernel_launch(void* const* d_in, const int* in_sizes, int n_in,
                              void* d_out, int out_size, void* d_ws, size_t ws_size,
                              hipStream_t stream) {
    (void)in_sizes; (void)n_in; (void)out_size; (void)ws_size;
    const float* x    = (const float*)d_in[0];
    // d_in[1] = mask (causal, handled analytically)
    const float* Wqkv = (const float*)d_in[2];
    const float* Wo   = (const float*)d_in[3];
    float* out = (float*)d_out;

    const int BS = 2 * 2048;      // B*S = 4096
    const int DM = 2048;
    const int QKV = 3 * DM;       // 6144

    char* ws = (char*)d_ws;
    size_t off = 0;
    _Float16* xh    = (_Float16*)(ws + off); off += (size_t)BS  * DM  * 2;
    _Float16* wqkvh = (_Float16*)(ws + off); off += (size_t)QKV * DM  * 2;
    _Float16* woh   = (_Float16*)(ws + off); off += (size_t)DM  * DM  * 2;
    _Float16* qkvh  = (_Float16*)(ws + off); off += (size_t)BS  * QKV * 2;
    _Float16* ctxh  = (_Float16*)(ws + off); off += (size_t)BS  * DM  * 2;

    cvt_f32_f16<<<2048, 256, 0, stream>>>(x,    xh,    BS * DM);
    cvt_f32_f16<<<2048, 256, 0, stream>>>(Wqkv, wqkvh, QKV * DM);
    cvt_f32_f16<<<2048, 256, 0, stream>>>(Wo,   woh,   DM * DM);

    // qkv = x @ W_qkv^T   [4096 x 6144]
    wmma_gemm_nt<false><<<dim3(QKV / GBN, BS / GBM), 256, 0, stream>>>(
        xh, wqkvh, (void*)qkvh, BS, QKV, DM);

    // flash attention -> ctx [4096 x 2048] f16
    flash_attn_kernel<<<dim3(2048 / 32, 16, 2), 256, 0, stream>>>(qkvh, ctxh);

    // out = ctx @ W_o^T   [4096 x 2048] f32
    wmma_gemm_nt<true><<<dim3(DM / GBN, BS / GBM), 256, 0, stream>>>(
        ctxh, woh, (void*)out, BS, DM, DM);
}
